// RoPEAttention_36266703847520
// MI455X (gfx1250) — matrix operand order repaired and checked by reading, not compiled, not run on hardware
//
#include <hip/hip_runtime.h>

// ---------------------------------------------------------------------------
// RoPE attention for MI455X (gfx1250, wave32, WMMA 16x16x32 bf16)
// B=2, N=2048, C=1024, H=16, D=64, HALF=32
// GEMMs: 128x64 block tile, 4 waves, 2 M-frags/wave (8 WMMA per wave per
// K-step), register-prefetch pipeline to overlap global loads with WMMA.
// ---------------------------------------------------------------------------

typedef __bf16 bf16_t;
typedef __attribute__((ext_vector_type(16))) __bf16 v16bf;
typedef __attribute__((ext_vector_type(8)))  __bf16 bf16x8;
typedef __attribute__((ext_vector_type(8)))  float  v8f;

#define Bdim 2
#define Nseq 2048
#define Cdim 1024
#define Hn   16
#define Dh   64

union FragBF { v16bf v; bf16x8 h[2]; };

__device__ inline v8f zero8() {
  v8f z;
  #pragma unroll
  for (int i = 0; i < 8; ++i) z[i] = 0.0f;
  return z;
}

__device__ inline bf16x8 pack8(float4 a, float4 b) {
  bf16x8 r = { (__bf16)a.x, (__bf16)a.y, (__bf16)a.z, (__bf16)a.w,
               (__bf16)b.x, (__bf16)b.y, (__bf16)b.z, (__bf16)b.w };
  return r;
}

// ---------------------------------------------------------------------------
// Kernel 1: qkv = x @ qkv_w.T  (M=4096, N'=3072, K=1024), RoPE epilogue.
//   q -> (B,H,N,D) bf16 (roped)     k -> (B,H,N,D) bf16 (roped)
//   v -> (B,H,D,N) bf16 (transposed so PV B-operand loads are contiguous)
// ---------------------------------------------------------------------------
__global__ void __launch_bounds__(128)
qkv_rope_kernel(const float* __restrict__ x, const float* __restrict__ w,
                const float* __restrict__ fcos, const float* __restrict__ fsin,
                bf16_t* __restrict__ qb, bf16_t* __restrict__ kb,
                bf16_t* __restrict__ vt)
{
  __shared__ bf16_t As[128][32];  // [m_local][k]
  __shared__ bf16_t Bs[64][32];   // [n_local][k]

  const int t    = threadIdx.x;
  const int lane = t & 31;
  const int wave = t >> 5;
  const int g    = lane >> 4;
  const int ln   = lane & 15;
  const int m_base = blockIdx.y * 128;
  const int n_base = blockIdx.x * 64;

  v8f acc[2][4];
  #pragma unroll
  for (int ms = 0; ms < 2; ++ms)
    #pragma unroll
    for (int nt = 0; nt < 4; ++nt) acc[ms][nt] = zero8();

  // staging assignments: A row = t (32 floats), B row = t>>1, 16-float seg
  const float* arow = x + (size_t)(m_base + t) * Cdim;
  const float* brow = w + (size_t)(n_base + (t >> 1)) * Cdim + (t & 1) * 16;

  float4 ax[8], bx[4];
  #pragma unroll
  for (int i = 0; i < 8; ++i) ax[i] = ((const float4*)arow)[i];
  #pragma unroll
  for (int i = 0; i < 4; ++i) bx[i] = ((const float4*)brow)[i];

  for (int k0 = 0; k0 < Cdim; k0 += 32) {
    __syncthreads();   // previous iteration's LDS consumers done
    {
      bf16_t* ad = &As[t][0];
      *(bf16x8*)(ad)      = pack8(ax[0], ax[1]);
      *(bf16x8*)(ad + 8)  = pack8(ax[2], ax[3]);
      *(bf16x8*)(ad + 16) = pack8(ax[4], ax[5]);
      *(bf16x8*)(ad + 24) = pack8(ax[6], ax[7]);
      bf16_t* bd = &Bs[t >> 1][(t & 1) * 16];
      *(bf16x8*)(bd)     = pack8(bx[0], bx[1]);
      *(bf16x8*)(bd + 8) = pack8(bx[2], bx[3]);
    }
    __syncthreads();   // staging visible

    if (k0 + 32 < Cdim) {  // prefetch next K tile while WMMAs run
      #pragma unroll
      for (int i = 0; i < 8; ++i) ax[i] = ((const float4*)(arow + k0 + 32))[i];
      #pragma unroll
      for (int i = 0; i < 4; ++i) bx[i] = ((const float4*)(brow + k0 + 32))[i];
    }

    FragBF a0, a1;
    a0.h[0] = *(const bf16x8*)&As[wave * 32 + ln][8 * g];
    a0.h[1] = *(const bf16x8*)&As[wave * 32 + ln][16 + 8 * g];
    a1.h[0] = *(const bf16x8*)&As[wave * 32 + 16 + ln][8 * g];
    a1.h[1] = *(const bf16x8*)&As[wave * 32 + 16 + ln][16 + 8 * g];

    #pragma unroll
    for (int nt = 0; nt < 4; ++nt) {
      FragBF b;
      b.h[0] = *(const bf16x8*)&Bs[nt * 16 + ln][8 * g];
      b.h[1] = *(const bf16x8*)&Bs[nt * 16 + ln][16 + 8 * g];
      acc[0][nt] = __builtin_amdgcn_wmma_f32_16x16x32_bf16(
          false, a0.v, false, b.v, (short)0, acc[0][nt], false, false);
      acc[1][nt] = __builtin_amdgcn_wmma_f32_16x16x32_bf16(
          false, a1.v, false, b.v, (short)0, acc[1][nt], false, false);
    }
  }

  // Epilogue: C/D layout — lane l, reg r -> row m = r + 8g, col n = ln.
  // RoPE pair (2i, 2i+1) = adjacent n = adjacent lane -> shfl_xor(1).
  #pragma unroll
  for (int ms = 0; ms < 2; ++ms) {
    #pragma unroll
    for (int nt = 0; nt < 4; ++nt) {
      const int n_global = n_base + nt * 16 + ln;
      const int section  = n_global >> 10;     // 0=q 1=k 2=v (block-uniform)
      const int cg       = n_global & 1023;
      const int h        = cg >> 6;
      const int d        = cg & 63;
      #pragma unroll
      for (int r = 0; r < 8; ++r) {
        float val   = acc[ms][nt][r];
        float other = __shfl_xor(val, 1, 32);  // pair partner (uniform call)
        const int mrow = m_base + wave * 32 + ms * 16 + r + 8 * g;
        const int bidx = mrow >> 11;
        const int npos = mrow & (Nseq - 1);
        if (section < 2) {
          const int i = d >> 1;
          float c = fcos[npos * (Dh / 2) + i];
          float s = fsin[npos * (Dh / 2) + i];
          float o = ((lane & 1) == 0) ? (val * c - other * s)
                                      : (other * s + val * c);
          size_t idx = ((size_t)(bidx * Hn + h) * Nseq + npos) * Dh + d;
          if (section == 0) qb[idx] = (bf16_t)o;
          else              kb[idx] = (bf16_t)o;
        } else {
          size_t idx = ((size_t)(bidx * Hn + h) * Dh + d) * Nseq + npos;
          vt[idx] = (bf16_t)val;
        }
      }
    }
  }
}

// ---------------------------------------------------------------------------
// Kernel 2: flash attention. One wave per (b, h, 16-row q tile).
// S = Q K^T * D^-0.5 -> online softmax -> O += P V, kv chunks of 32.
// ---------------------------------------------------------------------------
__global__ void __launch_bounds__(32)
attn_kernel(const bf16_t* __restrict__ qb, const bf16_t* __restrict__ kb,
            const bf16_t* __restrict__ vt, bf16_t* __restrict__ ao)
{
  __shared__ bf16_t P[16][32];    // probs tile: C/D layout -> A layout transpose

  const int lane = threadIdx.x & 31;
  const int g    = lane >> 4;
  const int ln   = lane & 15;
  const int qt   = blockIdx.x & 127;           // q tile
  const int bh   = blockIdx.x >> 7;            // b*H + h

  const bf16_t* qrow = qb + ((size_t)bh * Nseq + qt * 16 + ln) * Dh;
  FragBF qf[2];
  #pragma unroll
  for (int j = 0; j < 2; ++j) {
    qf[j].h[0] = *(const bf16x8*)(qrow + 32 * j + 8 * g);
    qf[j].h[1] = *(const bf16x8*)(qrow + 32 * j + 16 + 8 * g);
  }

  float mrow[8], lrow[8], corr[8];
  v8f o[4];
  #pragma unroll
  for (int r = 0; r < 8; ++r) { mrow[r] = -3.0e38f; lrow[r] = 0.0f; }
  #pragma unroll
  for (int dt = 0; dt < 4; ++dt) o[dt] = zero8();

  const bf16_t* kbase = kb + (size_t)bh * Nseq * Dh;
  const bf16_t* vbase = vt + (size_t)bh * Dh * Nseq;

  for (int kv0 = 0; kv0 < Nseq; kv0 += 32) {
    // ---- S = Q K^T for 32 kv rows (two 16-col subtiles, K-loop over D=64) ----
    v8f sc[2];
    #pragma unroll
    for (int nt = 0; nt < 2; ++nt) {
      sc[nt] = zero8();
      const bf16_t* krow = kbase + (size_t)(kv0 + nt * 16 + ln) * Dh;
      #pragma unroll
      for (int j = 0; j < 2; ++j) {
        FragBF kf;  // B operand: lane=col (kv row), halfs K(dim d) = 32j + 8g + e, 32j + 16 + 8g + e
        kf.h[0] = *(const bf16x8*)(krow + 32 * j + 8 * g);
        kf.h[1] = *(const bf16x8*)(krow + 32 * j + 16 + 8 * g);
        sc[nt] = __builtin_amdgcn_wmma_f32_16x16x32_bf16(
            false, qf[j].v, false, kf.v, (short)0, sc[nt], false, false);
      }
    }

    // ---- online softmax (row stats reduced across the 16-lane group) ----
    #pragma unroll
    for (int r = 0; r < 8; ++r) {
      float s0 = sc[0][r] * 0.125f;   // D^-0.5 = 1/8
      float s1 = sc[1][r] * 0.125f;
      float cm = fmaxf(s0, s1);
      cm = fmaxf(cm, __shfl_xor(cm, 1, 32));
      cm = fmaxf(cm, __shfl_xor(cm, 2, 32));
      cm = fmaxf(cm, __shfl_xor(cm, 4, 32));
      cm = fmaxf(cm, __shfl_xor(cm, 8, 32));
      float mnew = fmaxf(mrow[r], cm);
      corr[r] = __expf(mrow[r] - mnew);
      mrow[r] = mnew;
      float p0 = __expf(s0 - mnew);
      float p1 = __expf(s1 - mnew);
      float rs = p0 + p1;
      rs += __shfl_xor(rs, 1, 32);
      rs += __shfl_xor(rs, 2, 32);
      rs += __shfl_xor(rs, 4, 32);
      rs += __shfl_xor(rs, 8, 32);
      lrow[r] = lrow[r] * corr[r] + rs;
      // scatter P (C/D layout) to LDS row-major [qrow][kvcol]
      P[r + 8 * g][ln]      = (bf16_t)p0;
      P[r + 8 * g][16 + ln] = (bf16_t)p1;
    }
    #pragma unroll
    for (int dt = 0; dt < 4; ++dt)
      #pragma unroll
      for (int r = 0; r < 8; ++r) o[dt][r] *= corr[r];

    // LDS is in-order per wave; wait for the probability stores anyway.
    asm volatile("s_wait_dscnt 0" ::: "memory");

    // reload P in A-operand layout
    FragBF pf;
    pf.h[0] = *(const bf16x8*)&P[ln][8 * g];
    pf.h[1] = *(const bf16x8*)&P[ln][16 + 8 * g];

    // ---- O += P V : V^T layout makes B-operand loads contiguous ----
    #pragma unroll
    for (int dt = 0; dt < 4; ++dt) {
      FragBF vf;  // lane=col d=16dt+ln, halfs K(kv) = kv0 + 8g + e, kv0 + 16 + 8g + e
      const bf16_t* vrow = vbase + (size_t)(dt * 16 + ln) * Nseq + kv0 + 8 * g;
      vf.h[0] = *(const bf16x8*)(vrow);
      vf.h[1] = *(const bf16x8*)(vrow + 16);
      o[dt] = __builtin_amdgcn_wmma_f32_16x16x32_bf16(
          false, pf.v, false, vf.v, (short)0, o[dt], false, false);
    }
  }

  // normalize + store to (B*N, C) bf16 for the projection GEMM
  const int b = bh >> 4, h = bh & 15;
  #pragma unroll
  for (int dt = 0; dt < 4; ++dt) {
    #pragma unroll
    for (int r = 0; r < 8; ++r) {
      int m   = b * Nseq + qt * 16 + r + 8 * g;
      int col = h * Dh + dt * 16 + ln;
      ao[(size_t)m * Cdim + col] = (bf16_t)(o[dt][r] / lrow[r]);
    }
  }
}

// ---------------------------------------------------------------------------
// Kernel 3: out = attn_out @ proj_w.T + proj_b   (M=4096, N=1024, K=1024)
// Same 128x64 pipelined tile; A operand is already bf16.
// ---------------------------------------------------------------------------
__global__ void __launch_bounds__(128)
proj_kernel(const bf16_t* __restrict__ a, const float* __restrict__ w,
            const float* __restrict__ bias, float* __restrict__ out)
{
  __shared__ bf16_t As[128][32];
  __shared__ bf16_t Bs[64][32];

  const int t    = threadIdx.x;
  const int lane = t & 31;
  const int wave = t >> 5;
  const int g    = lane >> 4;
  const int ln   = lane & 15;
  const int m_base = blockIdx.y * 128;
  const int n_base = blockIdx.x * 64;

  v8f acc[2][4];
  #pragma unroll
  for (int ms = 0; ms < 2; ++ms)
    #pragma unroll
    for (int nt = 0; nt < 4; ++nt) acc[ms][nt] = zero8();

  const bf16_t* arow = a + (size_t)(m_base + t) * Cdim;
  const float*  brow = w + (size_t)(n_base + (t >> 1)) * Cdim + (t & 1) * 16;

  bf16x8 ax[4];
  float4 bx[4];
  #pragma unroll
  for (int i = 0; i < 4; ++i) ax[i] = ((const bf16x8*)arow)[i];
  #pragma unroll
  for (int i = 0; i < 4; ++i) bx[i] = ((const float4*)brow)[i];

  for (int k0 = 0; k0 < Cdim; k0 += 32) {
    __syncthreads();
    {
      bf16_t* ad = &As[t][0];
      *(bf16x8*)(ad)      = ax[0];
      *(bf16x8*)(ad + 8)  = ax[1];
      *(bf16x8*)(ad + 16) = ax[2];
      *(bf16x8*)(ad + 24) = ax[3];
      bf16_t* bd = &Bs[t >> 1][(t & 1) * 16];
      *(bf16x8*)(bd)     = pack8(bx[0], bx[1]);
      *(bf16x8*)(bd + 8) = pack8(bx[2], bx[3]);
    }
    __syncthreads();

    if (k0 + 32 < Cdim) {
      #pragma unroll
      for (int i = 0; i < 4; ++i) ax[i] = ((const bf16x8*)(arow + k0 + 32))[i];
      #pragma unroll
      for (int i = 0; i < 4; ++i) bx[i] = ((const float4*)(brow + k0 + 32))[i];
    }

    FragBF a0, a1;
    a0.h[0] = *(const bf16x8*)&As[wave * 32 + ln][8 * g];
    a0.h[1] = *(const bf16x8*)&As[wave * 32 + ln][16 + 8 * g];
    a1.h[0] = *(const bf16x8*)&As[wave * 32 + 16 + ln][8 * g];
    a1.h[1] = *(const bf16x8*)&As[wave * 32 + 16 + ln][16 + 8 * g];

    #pragma unroll
    for (int nt = 0; nt < 4; ++nt) {
      FragBF b;
      b.h[0] = *(const bf16x8*)&Bs[nt * 16 + ln][8 * g];
      b.h[1] = *(const bf16x8*)&Bs[nt * 16 + ln][16 + 8 * g];
      acc[0][nt] = __builtin_amdgcn_wmma_f32_16x16x32_bf16(
          false, a0.v, false, b.v, (short)0, acc[0][nt], false, false);
      acc[1][nt] = __builtin_amdgcn_wmma_f32_16x16x32_bf16(
          false, a1.v, false, b.v, (short)0, acc[1][nt], false, false);
    }
  }

  #pragma unroll
  for (int ms = 0; ms < 2; ++ms) {
    #pragma unroll
    for (int nt = 0; nt < 4; ++nt) {
      const int n_global = n_base + nt * 16 + ln;
      const float bn = bias[n_global];
      #pragma unroll
      for (int r = 0; r < 8; ++r) {
        const int mrow = m_base + wave * 32 + ms * 16 + r + 8 * g;
        out[(size_t)mrow * Cdim + n_global] = acc[ms][nt][r] + bn;
      }
    }
  }
}

// ---------------------------------------------------------------------------
extern "C" void kernel_launch(void* const* d_in, const int* in_sizes, int n_in,
                              void* d_out, int out_size, void* d_ws, size_t ws_size,
                              hipStream_t stream) {
  const float* x      = (const float*)d_in[0];
  const float* qkv_w  = (const float*)d_in[1];
  const float* proj_w = (const float*)d_in[2];
  const float* proj_b = (const float*)d_in[3];
  const float* fcos   = (const float*)d_in[4];
  const float* fsin   = (const float*)d_in[5];
  float* out = (float*)d_out;

  const size_t elems = (size_t)Bdim * Hn * Nseq * Dh;  // 4,194,304
  bf16_t* qb = (bf16_t*)d_ws;       // (B,H,N,D)   8 MB
  bf16_t* kb = qb + elems;          // (B,H,N,D)   8 MB
  bf16_t* vt = kb + elems;          // (B,H,D,N)   8 MB
  bf16_t* ao = vt + elems;          // (B*N, C)    8 MB

  dim3 g1(3 * Cdim / 64, (Bdim * Nseq) / 128);         // 48 x 32
  qkv_rope_kernel<<<g1, 128, 0, stream>>>(x, qkv_w, fcos, fsin, qb, kb, vt);

  attn_kernel<<<Bdim * Hn * (Nseq / 16), 32, 0, stream>>>(qb, kb, vt, ao);

  dim3 g3(Cdim / 64, (Bdim * Nseq) / 128);             // 16 x 32
  proj_kernel<<<g3, 128, 0, stream>>>(ao, proj_w, proj_b, out);
}
